// MaskConditionalTransformer_10445360464458
// MI455X (gfx1250) — compile-verified, matrix operand order checked by reading, not run on hardware
//
#include <hip/hip_runtime.h>

// ---------------------------------------------------------------------------
// MI455X / gfx1250, wave32. Matrix math via v_wmma_f32_16x16x32_bf16.
// Weights pre-converted to bf16; GEMM + flash-attention use double-buffered
// LDS tiles with software pipelining (global loads overlap WMMA).
// Epilogue variants are compile-time templated (no per-element branching).
// ---------------------------------------------------------------------------

typedef __attribute__((ext_vector_type(16))) __bf16 v16bf;
typedef __attribute__((ext_vector_type(8)))  float  v8f;

#define BB 8
#define SS 1024
#define EE 512
#define DD 1024
#define LL 8
#define HH 16
#define HD 64
#define FFN 4096
#define FUSED (3 * DD + 2 * FFN) // 11264
#define MTOT (BB * SS)           // 8192

__device__ __forceinline__ v8f wmma_bf16(v16bf a, v16bf b, v8f c) {
  return __builtin_amdgcn_wmma_f32_16x16x32_bf16(false, a, false, b,
                                                 (short)0, c, false, false);
}

// Load a 16-half WMMA fragment: two 16B chunks at p and p+16 halves.
__device__ __forceinline__ v16bf ldfrag(const __bf16* p) {
  union { uint4 u[2]; v16bf v; } r;
  r.u[0] = *(const uint4*)(p);
  r.u[1] = *(const uint4*)(p + 16);
  return r.v;
}

__device__ __forceinline__ float gelu_f(float x) {
  return x * 0.5f * (1.0f + erff(x * 0.70710678f));
}

// f32 -> bf16 bulk conversion (n multiple of 8).
__global__ __launch_bounds__(256) void k_cvt(const float* __restrict__ in,
                                             __bf16* __restrict__ out,
                                             long n) {
  const size_t i = ((size_t)blockIdx.x * 256 + threadIdx.x) * 8;
  if (i >= (size_t)n) return;
  float4 x = *(const float4*)(in + i);
  float4 y = *(const float4*)(in + i + 4);
  union { __bf16 h[8]; uint4 u; } pk;
  pk.h[0] = (__bf16)x.x; pk.h[1] = (__bf16)x.y;
  pk.h[2] = (__bf16)x.z; pk.h[3] = (__bf16)x.w;
  pk.h[4] = (__bf16)y.x; pk.h[5] = (__bf16)y.y;
  pk.h[6] = (__bf16)y.z; pk.h[7] = (__bf16)y.w;
  *(uint4*)(out + i) = pk.u;
}

// ---------------------------------------------------------------------------
// Time-embedding MLP (tiny): te = gelu([cos f, sin f]@t1^T+b1)@t2^T+b2
// ---------------------------------------------------------------------------
__global__ __launch_bounds__(256) void k_time(const float* __restrict__ times,
                                              const float* __restrict__ fw,
                                              const float* __restrict__ t1w,
                                              const float* __restrict__ t1b,
                                              const float* __restrict__ t2w,
                                              const float* __restrict__ t2b,
                                              float* __restrict__ te_out,
                                              float* __restrict__ silu_out) {
  __shared__ float s0[DD];
  __shared__ float s1[DD];
  const int b = blockIdx.x;
  const float t = times[b];
  for (int d = threadIdx.x; d < DD; d += 256) {
    float ang = 6.28318530718f * t * fw[(d < 512) ? d : (d - 512)];
    s0[d] = (d < 512) ? cosf(ang) : sinf(ang);
  }
  __syncthreads();
  for (int d = threadIdx.x; d < DD; d += 256) {
    const float* wr = t1w + (size_t)d * DD;
    float acc = t1b[d];
    for (int k = 0; k < DD; k++) acc += s0[k] * wr[k];
    s1[d] = gelu_f(acc);
  }
  __syncthreads();
  for (int d = threadIdx.x; d < DD; d += 256) {
    const float* wr = t2w + (size_t)d * DD;
    float acc = t2b[d];
    for (int k = 0; k < DD; k++) acc += s1[k] * wr[k];
    te_out[b * DD + d] = acc;
    silu_out[b * DD + d] = acc / (1.0f + __expf(-acc));
  }
}

// sb[l][b][n] = dot(silu_te[b], cond_w[l][n]) + cond_b[l][n]
__global__ __launch_bounds__(256) void k_cond(const float* __restrict__ st,
                                              const float* __restrict__ cw,
                                              const float* __restrict__ cb,
                                              float* __restrict__ sb) {
  const int idx = blockIdx.x * 256 + threadIdx.x;
  const int n = idx & (2 * DD - 1);
  const int rem = idx >> 11;
  const int b = rem & (BB - 1);
  const int l = rem >> 3;
  const float* s = st + b * DD;
  const float* w = cw + ((size_t)l * 2 * DD + n) * DD;
  float acc = cb[l * 2 * DD + n];
  for (int k = 0; k < DD; k++) acc += s[k] * w[k];
  sb[((size_t)l * BB + b) * 2 * DD + n] = acc;
}

// Build concat([noisy, cond, prev, infill*1]) as bf16 [M, 2048]
__global__ __launch_bounds__(256) void k_build_x(const float* __restrict__ ne,
                                                 const float* __restrict__ ce,
                                                 const float* __restrict__ pe,
                                                 const float* __restrict__ im,
                                                 __bf16* __restrict__ x16) {
  const size_t idx = (size_t)blockIdx.x * 256 + threadIdx.x;
  const int c = (int)(idx & 2047);
  const size_t m = idx >> 11;
  float v;
  if (c < 512)       v = ne[m * 512 + c];
  else if (c < 1024) v = ce[m * 512 + (c - 512)];
  else if (c < 1536) v = pe[m * 512 + (c - 1024)];
  else               v = im[m];
  x16[idx] = (__bf16)v;
}

// ---------------------------------------------------------------------------
// LayerNorm (+ optional scale/bias modulation, compile-time) -> bf16
// ---------------------------------------------------------------------------
__device__ __forceinline__ float block_sum_1024(float v, float* red) {
#pragma unroll
  for (int m = 16; m >= 1; m >>= 1) v += __shfl_xor(v, m, 32);
  const int wid = threadIdx.x >> 5;
  if ((threadIdx.x & 31) == 0) red[wid] = v;
  __syncthreads();
  float t = (threadIdx.x < 8) ? red[threadIdx.x] : 0.0f;
#pragma unroll
  for (int m = 4; m >= 1; m >>= 1) t += __shfl_xor(t, m, 32);
  if (threadIdx.x == 0) red[0] = t;
  __syncthreads();
  t = red[0];
  __syncthreads();
  return t;
}

template <bool MOD>
__global__ __launch_bounds__(256) void k_lnmod(const float* __restrict__ h,
                                               const float* __restrict__ lnw,
                                               const float* __restrict__ lnb,
                                               const float* __restrict__ sb,
                                               __bf16* __restrict__ out) {
  __shared__ float red[8];
  const int m = blockIdx.x;
  const int b = m >> 10;
  const float* row = h + (size_t)m * DD;
  float ls = 0.0f;
  float xv[4];
#pragma unroll
  for (int i = 0; i < 4; i++) {
    xv[i] = row[threadIdx.x + i * 256];
    ls += xv[i];
  }
  const float mu = block_sum_1024(ls, red) * (1.0f / DD);
  float lv = 0.0f;
#pragma unroll
  for (int i = 0; i < 4; i++) {
    float d = xv[i] - mu;
    lv += d * d;
  }
  const float var = block_sum_1024(lv, red) * (1.0f / DD);
  const float inv = rsqrtf(var + 1e-5f);
#pragma unroll
  for (int i = 0; i < 4; i++) {
    const int d = threadIdx.x + i * 256;
    float u = (xv[i] - mu) * inv * lnw[d] + lnb[d];
    if (MOD) {
      float sc = sb[b * 2 * DD + d];
      float bi = sb[b * 2 * DD + DD + d];
      u = (1.0f + sc) * u + bi;
    }
    out[(size_t)m * DD + d] = (__bf16)u;
  }
}

// Final LN over E=512 -> f32 output.
__global__ __launch_bounds__(128) void k_finln(const float* __restrict__ in,
                                               const float* __restrict__ w,
                                               const float* __restrict__ bvec,
                                               float* __restrict__ out) {
  __shared__ float red[4];
  const int m = blockIdx.x;
  const float* row = in + (size_t)m * EE;
  float xv[4], ls = 0.0f;
#pragma unroll
  for (int i = 0; i < 4; i++) {
    xv[i] = row[threadIdx.x + i * 128];
    ls += xv[i];
  }
#pragma unroll
  for (int msk = 16; msk >= 1; msk >>= 1) ls += __shfl_xor(ls, msk, 32);
  if ((threadIdx.x & 31) == 0) red[threadIdx.x >> 5] = ls;
  __syncthreads();
  const float mu = (red[0] + red[1] + red[2] + red[3]) * (1.0f / EE);
  __syncthreads();
  float lv = 0.0f;
#pragma unroll
  for (int i = 0; i < 4; i++) {
    float d = xv[i] - mu;
    lv += d * d;
  }
#pragma unroll
  for (int msk = 16; msk >= 1; msk >>= 1) lv += __shfl_xor(lv, msk, 32);
  if ((threadIdx.x & 31) == 0) red[threadIdx.x >> 5] = lv;
  __syncthreads();
  const float var = (red[0] + red[1] + red[2] + red[3]) * (1.0f / EE);
  const float inv = rsqrtf(var + 1e-5f);
#pragma unroll
  for (int i = 0; i < 4; i++) {
    const int d = threadIdx.x + i * 128;
    out[(size_t)m * EE + d] = (xv[i] - mu) * inv * w[d] + bvec[d];
  }
}

// ---------------------------------------------------------------------------
// RoPE + head split (q,k pre-scaled by HD^-0.5 each, matching reference).
// ---------------------------------------------------------------------------
__global__ __launch_bounds__(256) void k_rope(const __bf16* __restrict__ proj,
                                              __bf16* __restrict__ Q,
                                              __bf16* __restrict__ K,
                                              __bf16* __restrict__ V) {
  const size_t idx = (size_t)blockIdx.x * 256 + threadIdx.x;
  const int c = (int)(idx & 1023);
  const size_t m = idx >> 10;
  const int b = (int)(m >> 10), s = (int)(m & 1023);
  const int hh = c >> 6, d = c & 63;
  const size_t ro = m * FUSED;
  float q = (float)proj[ro + c];
  float k = (float)proj[ro + DD + c];
  float v = (float)proj[ro + 2 * DD + c];
  if (d < 32) {
    const int j = d & 15;
    const float invf = __powf(10000.0f, -(float)j / 16.0f);
    const float ang = (float)s * invf;
    const float cs = (float)(__bf16)cosf(ang);
    const float sn = (float)(__bf16)sinf(ang);
    const int pc = (d < 16) ? (c + 16) : (c - 16);
    float qp = (float)proj[ro + pc];
    float kp = (float)proj[ro + DD + pc];
    const float sgn = (d < 16) ? -1.0f : 1.0f;
    q = q * cs + sgn * qp * sn;
    k = k * cs + sgn * kp * sn;
  }
  const size_t o = (((size_t)(b * HH + hh)) * SS + s) * HD + d;
  Q[o] = (__bf16)(q * 0.125f);
  K[o] = (__bf16)(k * 0.125f);
  V[o] = (__bf16)v;
}

// GLU: g = ff * gelu(ff_gate)
__global__ __launch_bounds__(256) void k_glu(const __bf16* __restrict__ proj,
                                             __bf16* __restrict__ g16) {
  const size_t idx = (size_t)blockIdx.x * 256 + threadIdx.x;
  const int j = (int)(idx & (FFN - 1));
  const size_t m = idx >> 12;
  const size_t ro = m * FUSED + 3 * DD;
  float a = (float)proj[ro + j];
  float g = (float)proj[ro + FFN + j];
  g16[idx] = (__bf16)(a * gelu_f(g));
}

// ---------------------------------------------------------------------------
// GEMM: C[M,N] = A[M,K](bf16) @ W[N,K](bf16)^T. 128x128 block, 8 waves,
// wave tile 32x64, double-buffered LDS, pipelined global->LDS staging.
// Epilogue compile-time specialized: +bias, +add1, +add2; f32 or bf16 out.
// ---------------------------------------------------------------------------
template <bool HAS_BIAS, bool HAS_ADD1, bool HAS_ADD2, bool OUT_BF16>
__global__ __launch_bounds__(256) void k_gemm(const __bf16* __restrict__ A,
                                              const __bf16* __restrict__ Wb,
                                              const float* __restrict__ bias,
                                              const float* __restrict__ add1,
                                              const float* __restrict__ add2,
                                              float* __restrict__ outF,
                                              __bf16* __restrict__ outH,
                                              int N, int K) {
  __shared__ __bf16 As[2][128 * 40];
  __shared__ __bf16 Ws[2][128 * 40];
  const int tid = threadIdx.x;
  const int wid = tid >> 5, lane = tid & 31;
  const int l15 = lane & 15, hl = lane >> 4;
  const int m0 = blockIdx.y * 128;
  const int n0 = blockIdx.x * 128;
  const int wm = wid & 3;
  const int wn = wid >> 2;

  // Staging: thread owns 16B chunks (r, cc) and (r+64, cc).
  const int r = tid >> 2, cc = tid & 3;
  const size_t rowskip = (size_t)64 * K;
  const __bf16* aBase = A + (size_t)(m0 + r) * K + cc * 8;
  const __bf16* wBase = Wb + (size_t)(n0 + r) * K + cc * 8;
  const int soff = r * 40 + cc * 8;

  const v8f zf = {0.f, 0.f, 0.f, 0.f, 0.f, 0.f, 0.f, 0.f};
  v8f acc[2][4];
#pragma unroll
  for (int i = 0; i < 2; i++)
#pragma unroll
    for (int j = 0; j < 4; j++) acc[i][j] = zf;

  uint4 a0 = *(const uint4*)(aBase);
  uint4 a1 = *(const uint4*)(aBase + rowskip);
  uint4 w0 = *(const uint4*)(wBase);
  uint4 w1 = *(const uint4*)(wBase + rowskip);
  *(uint4*)(&As[0][soff]) = a0;
  *(uint4*)(&As[0][soff + 64 * 40]) = a1;
  *(uint4*)(&Ws[0][soff]) = w0;
  *(uint4*)(&Ws[0][soff + 64 * 40]) = w1;
  __syncthreads();

  const int nk = K >> 5;
  for (int kt = 0; kt < nk; kt++) {
    const int cur = kt & 1;
    const bool has = (kt + 1) < nk;
    if (has) {
      const __bf16* ap = aBase + (size_t)(kt + 1) * 32;
      const __bf16* wp = wBase + (size_t)(kt + 1) * 32;
      a0 = *(const uint4*)(ap);
      a1 = *(const uint4*)(ap + rowskip);
      w0 = *(const uint4*)(wp);
      w1 = *(const uint4*)(wp + rowskip);
    }
    v16bf af[2], bfr[4];
#pragma unroll
    for (int i = 0; i < 2; i++)
      af[i] = ldfrag(&As[cur][(wm * 32 + i * 16 + l15) * 40 + hl * 8]);
#pragma unroll
    for (int j = 0; j < 4; j++)
      bfr[j] = ldfrag(&Ws[cur][(wn * 64 + j * 16 + l15) * 40 + hl * 8]);
#pragma unroll
    for (int i = 0; i < 2; i++)
#pragma unroll
      for (int j = 0; j < 4; j++)
        acc[i][j] = wmma_bf16(af[i], bfr[j], acc[i][j]);
    if (has) {
      const int nb = cur ^ 1;
      *(uint4*)(&As[nb][soff]) = a0;
      *(uint4*)(&As[nb][soff + 64 * 40]) = a1;
      *(uint4*)(&Ws[nb][soff]) = w0;
      *(uint4*)(&Ws[nb][soff + 64 * 40]) = w1;
    }
    __syncthreads();
  }

  const int hi8 = (lane & 16) ? 8 : 0;
#pragma unroll
  for (int i = 0; i < 2; i++) {
#pragma unroll
    for (int j = 0; j < 4; j++) {
      const int n = n0 + wn * 64 + j * 16 + l15;
      const int mb = m0 + wm * 32 + i * 16 + hi8;
      const float bv = HAS_BIAS ? bias[n] : 0.0f;
#pragma unroll
      for (int rr = 0; rr < 8; rr++) {
        const size_t o = (size_t)(mb + rr) * N + n;
        float v = acc[i][j][rr] + bv;
        if (HAS_ADD1) v += add1[o];
        if (HAS_ADD2) v += add2[o];
        if (OUT_BF16) outH[o] = (__bf16)v;
        else          outF[o] = v;
      }
    }
  }
}

// ---------------------------------------------------------------------------
// Flash attention: block = (q-tile 64, head, batch), 4 waves x 16 q-rows.
// Double-buffered K/V tiles; QK^T and P.V via WMMA; online softmax.
// ---------------------------------------------------------------------------
__global__ __launch_bounds__(128) void k_attn(const __bf16* __restrict__ Q,
                                              const __bf16* __restrict__ K,
                                              const __bf16* __restrict__ V,
                                              const float* __restrict__ amask,
                                              __bf16* __restrict__ Out) {
  __shared__ __bf16 Kt[2][64 * 72];     // [key][d]
  __shared__ __bf16 Vt[2][64 * 72];     // [d][key] transposed
  __shared__ __bf16 Pt[4][16 * 72];     // per-wave P scratch
  const int tid = threadIdx.x;
  const int wid = tid >> 5, lane = tid & 31;
  const int l15 = lane & 15, hl = lane >> 4;
  const int hi8 = (lane & 16) ? 8 : 0;
  const int q0 = blockIdx.x * 64;
  const int hh = blockIdx.y, b = blockIdx.z;
  const size_t ho = ((size_t)(b * HH + hh)) * SS * HD;

  v16bf qf[2];
  {
    const __bf16* qp = Q + ho + (size_t)(q0 + wid * 16 + l15) * HD + hl * 8;
    qf[0] = ldfrag(qp);
    qf[1] = ldfrag(qp + 32);
  }
  const v8f zf = {0.f, 0.f, 0.f, 0.f, 0.f, 0.f, 0.f, 0.f};
  v8f o[4];
#pragma unroll
  for (int j = 0; j < 4; j++) o[j] = zf;
  float mrow[8], lrow[8];
#pragma unroll
  for (int rr = 0; rr < 8; rr++) { mrow[rr] = -1e30f; lrow[rr] = 0.0f; }
  const float* mb = amask + (size_t)b * SS;

  // Staging: thread owns key-row rk, 32-half segment seg of K and V.
  const int rk = tid & 63, seg = (tid >> 6) * 32;
  const __bf16* kBase = K + ho + (size_t)rk * HD + seg;
  const __bf16* vBase = V + ho + (size_t)rk * HD + seg;
  union VU { uint4 u[2]; __bf16 h[16]; };
  uint4 k0, k1, k2, k3, v0r, v1r, v2r, v3r;

  // Prologue: tile 0.
  k0 = *(const uint4*)(kBase);
  k1 = *(const uint4*)(kBase + 8);
  k2 = *(const uint4*)(kBase + 16);
  k3 = *(const uint4*)(kBase + 24);
  v0r = *(const uint4*)(vBase);
  v1r = *(const uint4*)(vBase + 8);
  v2r = *(const uint4*)(vBase + 16);
  v3r = *(const uint4*)(vBase + 24);
  {
    uint4* kd = (uint4*)(&Kt[0][rk * 72 + seg]);
    kd[0] = k0; kd[1] = k1; kd[2] = k2; kd[3] = k3;
    VU va, vb2;
    va.u[0] = v0r; va.u[1] = v1r;
    vb2.u[0] = v2r; vb2.u[1] = v3r;
#pragma unroll
    for (int d = 0; d < 16; d++) Vt[0][(seg + d) * 72 + rk] = va.h[d];
#pragma unroll
    for (int d = 0; d < 16; d++) Vt[0][(seg + 16 + d) * 72 + rk] = vb2.h[d];
  }
  __syncthreads();

  for (int t = 0; t < SS / 64; t++) {
    const int cur = t & 1;
    const bool has = (t + 1) < (SS / 64);
    if (has) {
      const size_t koff = (size_t)(t + 1) * 64 * HD;
      k0 = *(const uint4*)(kBase + koff);
      k1 = *(const uint4*)(kBase + koff + 8);
      k2 = *(const uint4*)(kBase + koff + 16);
      k3 = *(const uint4*)(kBase + koff + 24);
      v0r = *(const uint4*)(vBase + koff);
      v1r = *(const uint4*)(vBase + koff + 8);
      v2r = *(const uint4*)(vBase + koff + 16);
      v3r = *(const uint4*)(vBase + koff + 24);
    }

    // Scores: 4 tiles of 16q x 16k.
    v8f c[4];
#pragma unroll
    for (int j = 0; j < 4; j++) {
      const __bf16* kp = &Kt[cur][(j * 16 + l15) * 72 + hl * 8];
      v16bf b0 = ldfrag(kp);
      v16bf b1 = ldfrag(kp + 32);
      v8f cj = wmma_bf16(qf[0], b0, zf);
      cj = wmma_bf16(qf[1], b1, cj);
      const float mv = mb[t * 64 + j * 16 + l15];
#pragma unroll
      for (int rr = 0; rr < 8; rr++) cj[rr] += mv;
      c[j] = cj;
    }
    // Online softmax.
    float mt[8];
#pragma unroll
    for (int rr = 0; rr < 8; rr++)
      mt[rr] = fmaxf(fmaxf(c[0][rr], c[1][rr]), fmaxf(c[2][rr], c[3][rr]));
#pragma unroll
    for (int msk = 8; msk >= 1; msk >>= 1)
#pragma unroll
      for (int rr = 0; rr < 8; rr++)
        mt[rr] = fmaxf(mt[rr], __shfl_xor(mt[rr], msk, 32));
    float al[8];
#pragma unroll
    for (int rr = 0; rr < 8; rr++) {
      float mn = fmaxf(mrow[rr], mt[rr]);
      al[rr] = __expf(mrow[rr] - mn);
      mrow[rr] = mn;
    }
    float st[8];
#pragma unroll
    for (int rr = 0; rr < 8; rr++) st[rr] = 0.0f;
#pragma unroll
    for (int j = 0; j < 4; j++)
#pragma unroll
      for (int rr = 0; rr < 8; rr++) {
        float p = __expf(c[j][rr] - mrow[rr]);
        c[j][rr] = p;
        st[rr] += p;
      }
#pragma unroll
    for (int msk = 8; msk >= 1; msk >>= 1)
#pragma unroll
      for (int rr = 0; rr < 8; rr++) st[rr] += __shfl_xor(st[rr], msk, 32);
#pragma unroll
    for (int rr = 0; rr < 8; rr++) lrow[rr] = lrow[rr] * al[rr] + st[rr];
#pragma unroll
    for (int j = 0; j < 4; j++)
#pragma unroll
      for (int rr = 0; rr < 8; rr++) o[j][rr] *= al[rr];

    // P -> per-wave LDS region, reload as A fragments.
    __bf16* pw = &Pt[wid][0];
#pragma unroll
    for (int j = 0; j < 4; j++)
#pragma unroll
      for (int rr = 0; rr < 8; rr++)
        pw[(rr + hi8) * 72 + j * 16 + l15] = (__bf16)c[j][rr];
    __builtin_amdgcn_wave_barrier();
    const __bf16* pp = pw + l15 * 72 + hl * 8;
    v16bf pf0 = ldfrag(pp);
    v16bf pf1 = ldfrag(pp + 32);

    // O += P(16x64) @ V(64x64).
#pragma unroll
    for (int jd = 0; jd < 4; jd++) {
      const __bf16* vp = &Vt[cur][(jd * 16 + l15) * 72 + hl * 8];
      v16bf va = ldfrag(vp);
      v16bf vb2 = ldfrag(vp + 32);
      o[jd] = wmma_bf16(pf0, va, o[jd]);
      o[jd] = wmma_bf16(pf1, vb2, o[jd]);
    }

    if (has) {
      const int nb = cur ^ 1;
      uint4* kd = (uint4*)(&Kt[nb][rk * 72 + seg]);
      kd[0] = k0; kd[1] = k1; kd[2] = k2; kd[3] = k3;
      VU va, vb2;
      va.u[0] = v0r; va.u[1] = v1r;
      vb2.u[0] = v2r; vb2.u[1] = v3r;
#pragma unroll
      for (int d = 0; d < 16; d++) Vt[nb][(seg + d) * 72 + rk] = va.h[d];
#pragma unroll
      for (int d = 0; d < 16; d++) Vt[nb][(seg + 16 + d) * 72 + rk] = vb2.h[d];
    }
    __syncthreads();
  }

  // Normalize and write merged [b, s, h*64+d] as bf16.
#pragma unroll
  for (int jd = 0; jd < 4; jd++)
#pragma unroll
    for (int rr = 0; rr < 8; rr++) {
      const int qrow = q0 + wid * 16 + hi8 + rr;
      const int d = jd * 16 + l15;
      Out[((size_t)(b * SS + qrow)) * DD + hh * HD + d] =
          (__bf16)(o[jd][rr] / lrow[rr]);
    }
}

// ---------------------------------------------------------------------------
// Host launch
// ---------------------------------------------------------------------------
extern "C" void kernel_launch(void* const* d_in, const int* in_sizes, int n_in,
                              void* d_out, int out_size, void* d_ws, size_t ws_size,
                              hipStream_t stream) {
  const float* noisy  = (const float*)d_in[0];
  const float* conde  = (const float*)d_in[1];
  const float* preve  = (const float*)d_in[2];
  const float* infill = (const float*)d_in[3];
  const float* amask  = (const float*)d_in[4];
  const float* times  = (const float*)d_in[5];
  const float* fw     = (const float*)d_in[6];
  const float* t1w    = (const float*)d_in[7];
  const float* t1b    = (const float*)d_in[8];
  const float* t2w    = (const float*)d_in[9];
  const float* t2b    = (const float*)d_in[10];
  const float* in_w   = (const float*)d_in[11];
  const float* in_b   = (const float*)d_in[12];
  const float* ln_w   = (const float*)d_in[13];
  const float* ln_b   = (const float*)d_in[14];
  const float* cond_w = (const float*)d_in[15];
  const float* cond_b = (const float*)d_in[16];
  const float* proj_w = (const float*)d_in[17];
  const float* attn_w = (const float*)d_in[18];
  const float* attn_b = (const float*)d_in[19];
  const float* ff_w   = (const float*)d_in[20];
  const float* ff_b   = (const float*)d_in[21];
  const float* outlnw = (const float*)d_in[22];
  const float* outlnb = (const float*)d_in[23];
  const float* out_w  = (const float*)d_in[24];
  const float* out_b  = (const float*)d_in[25];
  const float* fin_w  = (const float*)d_in[26];
  const float* fin_b  = (const float*)d_in[27];
  float* outp = (float*)d_out;

  char* ws = (char*)d_ws;
  size_t off = 0;
  auto carve = [&](size_t bytes) { char* p = ws + off; off += (bytes + 255) & ~(size_t)255; return p; };
  float*  te     = (float*) carve(BB * DD * 4);
  float*  silu   = (float*) carve(BB * DD * 4);
  float*  sb     = (float*) carve((size_t)LL * BB * 2 * DD * 4);
  __bf16* x16    = (__bf16*)carve((size_t)MTOT * 2048 * 2);
  float*  h      = (float*) carve((size_t)MTOT * DD * 4);
  __bf16* u16    = (__bf16*)carve((size_t)MTOT * DD * 2);
  __bf16* p16    = (__bf16*)carve((size_t)MTOT * FUSED * 2);
  __bf16* Qb     = (__bf16*)carve((size_t)MTOT * DD * 2);
  __bf16* Kb     = (__bf16*)carve((size_t)MTOT * DD * 2);
  __bf16* Vb     = (__bf16*)carve((size_t)MTOT * DD * 2);
  __bf16* g16    = (__bf16*)carve((size_t)MTOT * FFN * 2);
  __bf16* am16   = (__bf16*)carve((size_t)MTOT * DD * 2);
  float*  tmp1   = (float*) carve((size_t)MTOT * DD * 4);
  float*  opre   = (float*) carve((size_t)MTOT * EE * 4);
  __bf16* w_in   = (__bf16*)carve((size_t)DD * 2048 * 2);
  __bf16* w_proj = (__bf16*)carve((size_t)LL * FUSED * DD * 2);
  __bf16* w_attn = (__bf16*)carve((size_t)LL * DD * DD * 2);
  __bf16* w_ff   = (__bf16*)carve((size_t)LL * DD * FFN * 2);
  __bf16* w_out  = (__bf16*)carve((size_t)EE * DD * 2);
  (void)ws_size; (void)in_sizes; (void)n_in; (void)out_size;

  // Weight conversion to bf16 (~0.8GB traffic once vs 2.5 TFLOP of math).
  {
    long n;
    n = (long)DD * 2048;
    k_cvt<<<(unsigned)(n / 2048), 256, 0, stream>>>(in_w, w_in, n);
    n = (long)LL * FUSED * DD;
    k_cvt<<<(unsigned)(n / 2048), 256, 0, stream>>>(proj_w, w_proj, n);
    n = (long)LL * DD * DD;
    k_cvt<<<(unsigned)(n / 2048), 256, 0, stream>>>(attn_w, w_attn, n);
    n = (long)LL * DD * FFN;
    k_cvt<<<(unsigned)(n / 2048), 256, 0, stream>>>(ff_w, w_ff, n);
    n = (long)EE * DD;
    k_cvt<<<(unsigned)(n / 2048), 256, 0, stream>>>(out_w, w_out, n);
  }

  k_time<<<BB, 256, 0, stream>>>(times, fw, t1w, t1b, t2w, t2b, te, silu);
  k_cond<<<(LL * BB * 2 * DD) / 256, 256, 0, stream>>>(silu, cond_w, cond_b, sb);

  k_build_x<<<(MTOT * 2048) / 256, 256, 0, stream>>>(noisy, conde, preve, infill, x16);
  k_gemm<true, false, false, false><<<dim3(DD / 128, MTOT / 128), 256, 0, stream>>>(
      x16, w_in, in_b, nullptr, nullptr, h, nullptr, DD, 4 * EE);

  for (int i = 0; i < LL; i++) {
    const float* sbi = sb + (size_t)i * BB * 2 * DD;
    k_lnmod<true><<<MTOT, 256, 0, stream>>>(h, ln_w + (size_t)i * DD,
                                            ln_b + (size_t)i * DD, sbi, u16);
    k_gemm<false, false, false, true><<<dim3(FUSED / 128, MTOT / 128), 256, 0, stream>>>(
        u16, w_proj + (size_t)i * FUSED * DD, nullptr, nullptr, nullptr,
        nullptr, p16, FUSED, DD);
    k_rope<<<(MTOT * DD) / 256, 256, 0, stream>>>(p16, Qb, Kb, Vb);
    k_attn<<<dim3(SS / 64, HH, BB), 128, 0, stream>>>(Qb, Kb, Vb, amask, am16);
    k_glu<<<(MTOT * FFN) / 256, 256, 0, stream>>>(p16, g16);
    k_gemm<true, false, false, false><<<dim3(DD / 128, MTOT / 128), 256, 0, stream>>>(
        am16, w_attn + (size_t)i * DD * DD, attn_b + (size_t)i * DD,
        nullptr, nullptr, tmp1, nullptr, DD, DD);
    // h = h + attn_out + ff_out (fused into FF epilogue)
    k_gemm<true, true, true, false><<<dim3(DD / 128, MTOT / 128), 256, 0, stream>>>(
        g16, w_ff + (size_t)i * DD * FFN, ff_b + (size_t)i * DD,
        h, tmp1, h, nullptr, DD, FFN);
  }

  k_lnmod<false><<<MTOT, 256, 0, stream>>>(h, outlnw, outlnb, nullptr, u16);
  k_gemm<true, false, false, false><<<dim3(EE / 128, MTOT / 128), 256, 0, stream>>>(
      u16, w_out, out_b, nullptr, nullptr, opre, nullptr, EE, DD);
  k_finln<<<MTOT, 128, 0, stream>>>(opre, fin_w, fin_b, outp);
}
